// TransformerAggregator_36472862277723
// MI455X (gfx1250) — matrix operand order repaired and checked by reading, not compiled, not run on hardware
//
#include <hip/hip_runtime.h>
#include <hip/hip_bf16.h>

#define NTOT 50000
#define KNB 20
#define DN 172
#define DE 172
#define DT 100
#define DOUT 128
#define DH 64
#define NHEAD 2
#define KQ 272       // DN+DT
#define KKV 444      // DN+DE+DT
#define KKVP 448     // padded to 14*32
#define KO 300       // DOUT+DN
#define NB 8         // nodes per block
#define ROWS (NB*KNB)    // 160 rows -> 10 M-tiles
#define ASTR 456         // A row stride (bf16 elems), bank-conflict-free
#define CSTR 260         // C row stride (f32 elems)
#define NCOLS 256        // combined [h_k | h_v] output columns

// LDS layout (bytes)
#define OFF_A   0
#define OFF_C   (ROWS*ASTR*2)                  // 145920
#define OFF_HQ  (OFF_C + ROWS*CSTR*4)          // 312320
#define OFF_HN  (OFF_HQ + NB*DOUT*4)           // 316416
#define OFF_ATT (OFF_HN + NB*DOUT*4)           // 320512
#define OFF_ZTF (OFF_ATT + NB*NHEAD*KNB*4)     // 321792
#define SMEM_BYTES (OFF_ZTF + DT*4)            // 322192 < 327680

typedef __attribute__((ext_vector_type(16))) __bf16 v16bf;
typedef __attribute__((ext_vector_type(8)))  float  v8f;

static __device__ __forceinline__ unsigned short f2bf(float x) {
  union { float f; unsigned u; } v; v.f = x;
  unsigned r = v.u + 0x7FFFu + ((v.u >> 16) & 1u);   // round-to-nearest-even
  return (unsigned short)(r >> 16);
}

// ---- Kernel 0: pack wk|wv into bf16 B matrix [256][448] (row n = output col) ----
__global__ void prep_wkv(const float* __restrict__ wkW, const float* __restrict__ wvW,
                         unsigned short* __restrict__ wkv) {
  int idx = blockIdx.x * blockDim.x + threadIdx.x;
  if (idx >= NCOLS * KKVP) return;
  int n = idx / KKVP, k = idx - n * KKVP;
  float v = 0.f;
  if (k < KKV) v = (n < DOUT) ? wkW[(size_t)n * KKV + k] : wvW[(size_t)(n - DOUT) * KKV + k];
  wkv[idx] = f2bf(v);
}

// ---- Main kernel: 8 nodes per block, 256 threads (8 wave32) ----
__global__ __launch_bounds__(256) void tga_main(
    const float* __restrict__ rootF,  const float* __restrict__ rootTs,
    const float* __restrict__ nbrTs,  const float* __restrict__ nbrF,
    const float* __restrict__ edgeF,  const float* __restrict__ timeW,
    const float* __restrict__ timeB,  const float* __restrict__ wqW,
    const float* __restrict__ wqB,    const unsigned short* __restrict__ wkv,
    const float* __restrict__ wkB,    const float* __restrict__ wvB,
    const float* __restrict__ woW,    const float* __restrict__ woB,
    const float* __restrict__ lnG,    const float* __restrict__ lnB,
    float* __restrict__ out)
{
  extern __shared__ char smem[];
  const int tid  = threadIdx.x;
  const int lane = tid & 31;
  const int wave = tid >> 5;
  const int g0   = blockIdx.x * NB;

  unsigned short* A = (unsigned short*)(smem + OFF_A);
  float* C   = (float*)(smem + OFF_C);
  float* HQ  = (float*)(smem + OFF_HQ);
  float* HN  = (float*)(smem + OFF_HN);
  float* ATT = (float*)(smem + OFF_ATT);
  float* ZTF = (float*)(smem + OFF_ZTF);

  // ---------- Phase 1: build bf16 kv_in A-matrix in LDS ----------
  for (int idx = tid; idx < ROWS * (KKVP / 4); idx += 256) {
    int r   = idx / (KKVP / 4);
    int cg  = idx - r * (KKVP / 4);
    int col = cg * 4;
    int i = r / KNB, j = r - i * KNB;
    int g = g0 + i;
    float v0 = 0.f, v1 = 0.f, v2 = 0.f, v3 = 0.f;
    if (col < DN) {
      const float* p = nbrF + ((size_t)g * KNB + j) * DN + col;
      v0 = p[0]; v1 = p[1]; v2 = p[2]; v3 = p[3];
    } else if (col < DN + DE) {
      const float* p = edgeF + ((size_t)g * KNB + j) * DE + (col - DN);
      v0 = p[0]; v1 = p[1]; v2 = p[2]; v3 = p[3];
    } else if (col < KKV) {
      float dt = rootTs[g] - nbrTs[(size_t)g * KNB + j];
      int d = col - (DN + DE);
      v0 = cosf(dt * timeW[d + 0] + timeB[d + 0]);
      v1 = cosf(dt * timeW[d + 1] + timeB[d + 1]);
      v2 = cosf(dt * timeW[d + 2] + timeB[d + 2]);
      v3 = cosf(dt * timeW[d + 3] + timeB[d + 3]);
    }
    unsigned lo = (unsigned)f2bf(v0) | ((unsigned)f2bf(v1) << 16);
    unsigned hi = (unsigned)f2bf(v2) | ((unsigned)f2bf(v3) << 16);
    uint2 pk; pk.x = lo; pk.y = hi;
    *(uint2*)(A + (size_t)r * ASTR + col) = pk;   // 8B aligned ds_store_b64
  }
  // stage root features (flat 8x172 f32) into (not-yet-used) C region, and zero_tf
  for (int idx = tid; idx < NB * DN; idx += 256)
    C[idx] = rootF[(size_t)g0 * DN + idx];
  if (tid < DT) ZTF[tid] = cosf(timeB[tid]);
  __syncthreads();

  // ---------- Phase 1b: h_q (fp32, 8x128 outputs, K=272) ----------
  {
    int n  = tid & 127;
    int i0 = tid >> 7;                      // 0 or 1
    const float* wrow = wqW + (size_t)n * KQ;
    float a0 = wqB[n], a1 = a0, a2 = a0, a3 = a0;
    const float* r0 = C + (i0 + 0) * DN;
    const float* r1 = C + (i0 + 2) * DN;
    const float* r2 = C + (i0 + 4) * DN;
    const float* r3 = C + (i0 + 6) * DN;
    for (int c = 0; c < DN; ++c) {
      float w = wrow[c];
      a0 += w * r0[c]; a1 += w * r1[c]; a2 += w * r2[c]; a3 += w * r3[c];
    }
    for (int c = 0; c < DT; ++c) {
      float w = wrow[DN + c] * ZTF[c];
      a0 += w; a1 += w; a2 += w; a3 += w;
    }
    HQ[(i0 + 0) * DOUT + n] = a0; HQ[(i0 + 2) * DOUT + n] = a1;
    HQ[(i0 + 4) * DOUT + n] = a2; HQ[(i0 + 6) * DOUT + n] = a3;
  }
  __syncthreads();

  // ---------- Phase 2: WMMA GEMM -> C[160][256] = [h_k | h_v] ----------
  {
    const int l16 = lane & 15;
    const int hi  = lane >> 4;
    for (int it = 0; it < 20; ++it) {
      int job = it * 8 + wave;              // 160 tile jobs = 10 M x 16 N
      int mt = job >> 4, nt = job & 15;
      int m = mt * 16 + l16;
      int n = nt * 16 + l16;
      const unsigned short* arow = A   + (size_t)m * ASTR;
      const unsigned short* brow = wkv + (size_t)n * KKVP;
      v8f acc = {0.f, 0.f, 0.f, 0.f, 0.f, 0.f, 0.f, 0.f};
      #pragma unroll
      for (int ks = 0; ks < 14; ++ks) {
        int kb = ks * 32;
        union { uint4 q[2]; v16bf v; } av, bv;
        // A 16x32 bf16 layout: lane<16 -> K kb..kb+7 & kb+16..23; lane>=16 -> +8
        av.q[0] = *(const uint4*)(arow + kb + hi * 8);
        av.q[1] = *(const uint4*)(arow + kb + 16 + hi * 8);
        // B 32x16 bf16 layout: lane<16 -> K kb..kb+7 & kb+16..23; lane>=16 -> +8
        bv.q[0] = *(const uint4*)(brow + kb + hi * 8);
        bv.q[1] = *(const uint4*)(brow + kb + 16 + hi * 8);
        acc = __builtin_amdgcn_wmma_f32_16x16x32_bf16(
            false, av.v, false, bv.v, (short)0, acc, false, false);
      }
      float bias = (n < DOUT) ? wkB[n] : wvB[n - DOUT];
      int mbase = mt * 16 + hi * 8;         // C/D layout: VGPR v -> M = v (+8 for hi lanes)
      #pragma unroll
      for (int v = 0; v < 8; ++v)
        C[(size_t)(mbase + v) * CSTR + n] = acc[v] + bias;
    }
  }
  __syncthreads();

  // ---------- Phase 4a: attention scores (dot over dh=64) + LeakyReLU ----------
  for (int t = tid; t < NB * NHEAD * KNB; t += 256) {
    int i = t / (NHEAD * KNB);
    int rem = t - i * (NHEAD * KNB);
    int h = rem / KNB, j = rem - h * KNB;
    const float* q = HQ + i * DOUT + h * DH;
    const float* k = C + (size_t)(i * KNB + j) * CSTR + h * DH;
    float s = 0.f;
    for (int d = 0; d < DH; ++d) s += q[d] * k[d];
    s = (s > 0.f) ? s : 0.2f * s;
    ATT[(i * NHEAD + h) * KNB + j] = s;
  }
  __syncthreads();
  // ---------- Phase 4b: softmax over K=20 per (node, head) ----------
  if (tid < NB * NHEAD) {
    float* a = ATT + tid * KNB;
    float mx = a[0];
    for (int j = 1; j < KNB; ++j) mx = fmaxf(mx, a[j]);
    float sum = 0.f;
    for (int j = 0; j < KNB; ++j) { float e = expf(a[j] - mx); a[j] = e; sum += e; }
    float inv = 1.f / sum;
    for (int j = 0; j < KNB; ++j) a[j] *= inv;
  }
  __syncthreads();
  // ---------- Phase 4c: h_neigh = sum_j att * h_v ----------
  #pragma unroll
  for (int p = 0; p < 4; ++p) {
    int o = tid + p * 256;
    int i = o >> 7, d = o & 127;
    const float* a = ATT + (i * NHEAD + (d >> 6)) * KNB;
    const float* v = C + (size_t)(i * KNB) * CSTR + DOUT + d;
    float s = 0.f;
    for (int j = 0; j < KNB; ++j) s += a[j] * v[(size_t)j * CSTR];
    HN[o] = s;
  }
  __syncthreads();

  // ---------- Phase 5: output proj + ReLU (A region reused) ----------
  float* HO = (float*)(smem + OFF_A);
  #pragma unroll
  for (int p = 0; p < 4; ++p) {
    int o = tid + p * 256;
    int i = o >> 7, n = o & 127;
    const float* wrow = woW + (size_t)n * KO;
    float acc = woB[n];
    const float* hn = HN + i * DOUT;
    for (int c = 0; c < DOUT; ++c) acc += wrow[c] * hn[c];
    const float* rf = rootF + (size_t)(g0 + i) * DN;
    for (int c = 0; c < DN; ++c) acc += wrow[DOUT + c] * rf[c];
    HO[o] = fmaxf(acc, 0.f);
  }
  __syncthreads();
  if (tid < NB) {
    const float* h = HO + tid * DOUT;
    float s = 0.f, ss = 0.f;
    for (int c = 0; c < DOUT; ++c) { float x = h[c]; s += x; ss += x * x; }
    float mu = s * (1.f / DOUT);
    float var = ss * (1.f / DOUT) - mu * mu;
    ATT[tid]      = mu;
    ATT[NB + tid] = rsqrtf(var + 1e-5f);
  }
  __syncthreads();
  #pragma unroll
  for (int p = 0; p < 4; ++p) {
    int o = tid + p * 256;
    int i = o >> 7, n = o & 127;
    float y = (HO[o] - ATT[i]) * ATT[NB + i] * lnG[n] + lnB[n];
    out[(size_t)(g0 + i) * DOUT + n] = y;
  }
}

extern "C" void kernel_launch(void* const* d_in, const int* in_sizes, int n_in,
                              void* d_out, int out_size, void* d_ws, size_t ws_size,
                              hipStream_t stream) {
  const float* rootF  = (const float*)d_in[0];
  const float* rootTs = (const float*)d_in[1];
  const float* nbrTs  = (const float*)d_in[2];
  const float* nbrF   = (const float*)d_in[3];
  const float* edgeF  = (const float*)d_in[4];
  const float* timeW  = (const float*)d_in[5];
  const float* timeB  = (const float*)d_in[6];
  const float* wqW    = (const float*)d_in[7];
  const float* wqB    = (const float*)d_in[8];
  const float* wkW    = (const float*)d_in[9];
  const float* wkB    = (const float*)d_in[10];
  const float* wvW    = (const float*)d_in[11];
  const float* wvB    = (const float*)d_in[12];
  const float* woW    = (const float*)d_in[13];
  const float* woB    = (const float*)d_in[14];
  const float* lnG    = (const float*)d_in[15];
  const float* lnB    = (const float*)d_in[16];
  float* out = (float*)d_out;
  unsigned short* wkv = (unsigned short*)d_ws;   // 256*448*2 = 229376 B of scratch

  (void)hipFuncSetAttribute((const void*)tga_main,
                            hipFuncAttributeMaxDynamicSharedMemorySize, SMEM_BYTES);

  prep_wkv<<<(NCOLS * KKVP + 255) / 256, 256, 0, stream>>>(wkW, wvW, wkv);
  tga_main<<<NTOT / NB, 256, SMEM_BYTES, stream>>>(
      rootF, rootTs, nbrTs, nbrF, edgeF, timeW, timeB,
      wqW, wqB, wkv, wkB, wvB, woW, woB, lnG, lnB, out);
}
